// SingleHeadSelfAttention_81088982549088
// MI455X (gfx1250) — matrix operand order repaired and checked by reading, not compiled, not run on hardware
//
#include <hip/hip_runtime.h>
#include <math.h>

// ---------------------------------------------------------------------------
// Single-head causal self-attention for MI455X (gfx1250, wave32, WMMA).
//
// Roofline: attention matmuls ~140 GFLOP vs ~60 MB of traffic -> strongly
// compute bound; both big matmuls run on v_wmma_f32_16x16x32_f16 (f16 in,
// f32 accum).  The 24 MB f16 Q/K/V intermediate lives in d_ws and fits in
// the 192 MB L2, so the flash-loop K/V re-reads never touch HBM.
//
// New this round: K/V blocks are streamed into double-buffered LDS with the
// CDNA5 async engine (global_load_async_to_lds_b128 + s_wait_asynccnt), so
// the next 32-key block loads while the current block's WMMAs and softmax
// execute.  V is pre-transposed into a blocked [d][s] layout by the
// projection kernel so both K and V blocks are contiguous 4 KB async copies.
// ---------------------------------------------------------------------------

typedef __attribute__((ext_vector_type(16))) _Float16 v16h;
typedef __attribute__((ext_vector_type(8)))  _Float16 v8h;
typedef __attribute__((ext_vector_type(8)))  float    v8f;

union V16 { v16h v; v8h h[2]; };

static __device__ __forceinline__ v8f wmma_f16(v16h a, v16h b, v8f c) {
  // D = A(16x32 f16) x B(32x16 f16) + C(16x16 f32)
  return __builtin_amdgcn_wmma_f32_16x16x32_f16(
      /*neg_a=*/false, a, /*neg_b=*/false, b,
      /*c_mod=*/(short)0, c, /*reuse_a=*/false, /*reuse_b=*/false);
}

// Async-copy 4 KB (contiguous) from global to LDS: each of 128 threads moves
// 32 B via two b128 async ops.  LDS addr = VDST + offset, global addr =
// SADDR + VADDR + offset; tracked by ASYNCcnt.
static __device__ __forceinline__ void async_copy_4k(unsigned ldsBase,
                                                     const void* gsrc,
                                                     int tid) {
  const unsigned off = (unsigned)tid * 32u;
  const unsigned long long base = (unsigned long long)gsrc;
  asm volatile(
      "global_load_async_to_lds_b128 %0, %1, %2\n\t"
      "global_load_async_to_lds_b128 %0, %1, %2 offset:16"
      :
      : "v"(ldsBase + off), "v"(off), "s"(base)
      : "memory");
}

#define BATCH 16
#define SEQ   4096
#define DM    64

// ===========================================================================
// Kernel 1: QKV projection.  out = X @ W^T, per-WG tile of 64 rows.
//  - X (64x64 f32) and Wq/Wk/Wv (64x64 f32) staged in LDS as f16.
//  - A-frag (16x32 f16): lane l holds row (l&15); K elems
//      c = kb*32 + (e<8?0:16) + (l>=16?8:0) + (e&7)  -> two 16B chunks.
//  - B-frag (32x16 f16): lane l holds col (l&15); K elems
//      c = kb*32 + (e<8?0:16) + (l>=16?8:0) + (e&7)  -> two 16B chunks.
//  - D-frag (16x16 f32): VGPR r -> row r + 8*(l>=16), col (l&15).
// Q is pre-scaled by 1/sqrt(64) = 0.125.
// V is written blocked-transposed: [flat_row>>5][64 d][32 s_local] so each
// attention key-block is one contiguous 4 KB run for the async engine.
// ===========================================================================
__global__ __launch_bounds__(128) void qkv_proj_kernel(
    const float* __restrict__ x,
    const float* __restrict__ Wk,
    const float* __restrict__ Wq,
    const float* __restrict__ Wv,
    _Float16* __restrict__ Qh,
    _Float16* __restrict__ Kh,
    _Float16* __restrict__ Vh)
{
  __shared__ _Float16 Xh[64 * DM];
  __shared__ _Float16 Wh[3][64 * DM];

  const int tid  = threadIdx.x;
  const int lane = tid & 31;
  const int wave = tid >> 5;
  const int half = lane >> 4;   // 0: lanes 0-15, 1: lanes 16-31
  const int ln   = lane & 15;

  const size_t row0 = (size_t)blockIdx.x * 64;   // flat row over [B*T)

  // cooperative, coalesced load of the 64-row X tile (f32 -> f16)
  {
    const float2* src = (const float2*)(x + row0 * DM);
    for (int i = tid; i < 64 * DM / 2; i += 128) {
      float2 w = src[i];
      Xh[2 * i]     = (_Float16)w.x;
      Xh[2 * i + 1] = (_Float16)w.y;
    }
  }
  // weights: sel 0=Q, 1=K, 2=V  (note input order is Wk, Wq, Wv!)
  {
    const float* Ws[3] = {Wq, Wk, Wv};
    for (int s = 0; s < 3; s++) {
      const float2* src = (const float2*)Ws[s];
      for (int i = tid; i < 64 * DM / 2; i += 128) {
        float2 w = src[i];
        Wh[s][2 * i]     = (_Float16)w.x;
        Wh[s][2 * i + 1] = (_Float16)w.y;
      }
    }
  }
  __syncthreads();

  // A fragments for this wave's 16 rows (K = 64 -> 2 fragments)
  V16 a[2];
#pragma unroll
  for (int kb = 0; kb < 2; kb++) {
    const _Float16* p = &Xh[(wave * 16 + ln) * DM + kb * 32 + half * 8];
    a[kb].h[0] = *(const v8h*)p;          // c = kb*32 + half*8      .. +7
    a[kb].h[1] = *(const v8h*)(p + 16);   // c = kb*32 + 16 + half*8 .. +7
  }

  _Float16* outs[3] = {Qh, Kh, Vh};
#pragma unroll
  for (int s = 0; s < 3; s++) {
    const float scale = (s == 0) ? 0.125f : 1.0f;  // fold 1/sqrt(C) into Q
#pragma unroll
    for (int nt = 0; nt < 4; nt++) {               // output cols nt*16..+15
      v8f acc = {0.f, 0.f, 0.f, 0.f, 0.f, 0.f, 0.f, 0.f};
#pragma unroll
      for (int kb = 0; kb < 2; kb++) {
        V16 bf;   // B[c][n] = W[n_glob][c], row-major W -> contiguous in c
        const _Float16* p = &Wh[s][(nt * 16 + ln) * DM + kb * 32 + half * 8];
        bf.h[0] = *(const v8h*)p;
        bf.h[1] = *(const v8h*)(p + 16);
        acc = wmma_f16(a[kb].v, bf.v, acc);
      }
      _Float16* dst = outs[s];
#pragma unroll
      for (int r = 0; r < 8; r++) {
        const size_t grow = row0 + (size_t)(wave * 16 + r + half * 8);
        const int d = nt * 16 + ln;
        const _Float16 hv = (_Float16)(acc[r] * scale);
        if (s == 2) {
          // blocked transpose: [grow>>5][d][grow&31]
          dst[((grow >> 5) * DM + d) * 32 + (grow & 31)] = hv;
        } else {
          dst[grow * DM + d] = hv;
        }
      }
    }
  }
}

// ===========================================================================
// Kernel 2: flash attention (causal, online softmax), async double-buffered.
// WG = 128 threads (4 waves); WG owns 64 query rows of one batch; each wave
// owns a 16-row query tile.  Loop over 32-key blocks (causally truncated):
//   prefetch block sb+1 into buf^1 via async global->LDS (ASYNCcnt),
//   S(16x32)  = Q(16x64) @ K^T   : 4 WMMAs
//   online softmax (shfl_xor reductions within 16-lane halves)
//   y(16x64) += P(16x32) @ V     : 4 WMMAs
// P is relaid out D->A through a 1 KB per-wave LDS bounce (same-wave LDS is
// in-order).  V arrives already transposed ([d][s]) from the proj kernel.
// ===========================================================================
__global__ __launch_bounds__(128) void attn_kernel(
    const _Float16* __restrict__ Qh,
    const _Float16* __restrict__ Kh,
    const _Float16* __restrict__ Vh,
    float* __restrict__ out)
{
  __shared__ _Float16 Klds[2][32 * DM];   // K block, row-major [s][d]
  __shared__ _Float16 Vt[2][DM * 32];     // V block, transposed [d][s]
  __shared__ _Float16 Plds[4][16 * 32];   // per-wave P bounce buffer

  const int tid  = threadIdx.x;
  const int lane = tid & 31;
  const int wave = tid >> 5;
  const int half = lane >> 4;
  const int ln   = lane & 15;

  const int b  = blockIdx.x >> 6;         // T/64 = 64 q-blocks per batch
  const int q0 = (blockIdx.x & 63) * 64;  // WG query-block base
  const int qlo = q0 + wave * 16;         // this wave's first query row

  const unsigned kldsBase[2] = {
      (unsigned)(unsigned long long)(const void*)&Klds[0][0],
      (unsigned)(unsigned long long)(const void*)&Klds[1][0]};
  const unsigned vldsBase[2] = {
      (unsigned)(unsigned long long)(const void*)&Vt[0][0],
      (unsigned)(unsigned long long)(const void*)&Vt[1][0]};

  // --- load this wave's Q A-fragments (already scaled by 1/sqrt(C)) ---
  const size_t qbase = ((size_t)b * SEQ + qlo) * DM;
  V16 qa[2];
#pragma unroll
  for (int kb = 0; kb < 2; kb++) {
    const _Float16* p = Qh + qbase + (size_t)ln * DM + kb * 32 + half * 8;
    qa[kb].h[0] = *(const v8h*)p;
    qa[kb].h[1] = *(const v8h*)(p + 16);
  }

  v8f yacc[4];
  float rowmax[8], rowsum[8];
#pragma unroll
  for (int nt = 0; nt < 4; nt++) {
    v8f z = {0.f, 0.f, 0.f, 0.f, 0.f, 0.f, 0.f, 0.f};
    yacc[nt] = z;
  }
#pragma unroll
  for (int r = 0; r < 8; r++) { rowmax[r] = -INFINITY; rowsum[r] = 0.f; }

  const int nblk = (q0 + 64) / 32;        // causal truncation of key blocks

  // prologue: kick off block 0 into buffer 0 (ASYNCcnt batch of 4/thread)
  async_copy_4k(kldsBase[0], Kh + ((size_t)b * SEQ) * DM, tid);
  async_copy_4k(vldsBase[0], Vh + ((size_t)b * (SEQ / 32)) * DM * 32, tid);

  for (int sb = 0; sb < nblk; sb++) {
    const int s0  = sb * 32;
    const int cur = sb & 1;

    // all waves are done reading buf^1 (iteration sb-1) before we overwrite
    __syncthreads();
    if (sb + 1 < nblk) {
      const int nx = sb + 1;
      async_copy_4k(kldsBase[cur ^ 1],
                    Kh + ((size_t)b * SEQ + (size_t)nx * 32) * DM, tid);
      async_copy_4k(vldsBase[cur ^ 1],
                    Vh + ((size_t)(b * (SEQ / 32) + nx)) * DM * 32, tid);
      // batch for buf[cur] (4 older ops) retired once <=4 remain outstanding
      asm volatile("s_wait_asynccnt 0x4" ::: "memory");
    } else {
      asm volatile("s_wait_asynccnt 0x0" ::: "memory");
    }
    __syncthreads();   // everyone's batch for buf[cur] is in LDS

    // wave-uniform causal skip: block entirely above this wave's rows
    if (s0 <= qlo + 15) {
      const _Float16* Kb = &Klds[cur][0];
      const _Float16* Vb = &Vt[cur][0];
      // --- scores: S = Q @ K^T  (B[c][n] = K[s_loc][c], contiguous c) ---
      v8f sc[2];
#pragma unroll
      for (int nt = 0; nt < 2; nt++) {
        v8f acc = {0.f, 0.f, 0.f, 0.f, 0.f, 0.f, 0.f, 0.f};
#pragma unroll
        for (int kb = 0; kb < 2; kb++) {
          V16 bf;
          const _Float16* p = Kb + (nt * 16 + ln) * DM + kb * 32 + half * 8;
          bf.h[0] = *(const v8h*)p;
          bf.h[1] = *(const v8h*)(p + 16);
          acc = wmma_f16(qa[kb].v, bf.v, acc);
        }
        sc[nt] = acc;
      }
      // --- causal mask (element granularity) ---
#pragma unroll
      for (int nt = 0; nt < 2; nt++)
#pragma unroll
        for (int r = 0; r < 8; r++) {
          const int qg = qlo + r + half * 8;
          const int sg = s0 + nt * 16 + ln;
          if (sg > qg) sc[nt][r] = -INFINITY;
        }
      // --- online softmax update ---
#pragma unroll
      for (int r = 0; r < 8; r++) {
        float mx = fmaxf(sc[0][r], sc[1][r]);
#pragma unroll
        for (int off = 1; off < 16; off <<= 1)
          mx = fmaxf(mx, __shfl_xor(mx, off, 32));    // stays in lane half
        const float nm   = fmaxf(rowmax[r], mx);      // finite after block 0
        const float corr = __expf(rowmax[r] - nm);    // exp(-inf)=0 first time
        float ps = 0.f;
#pragma unroll
        for (int nt = 0; nt < 2; nt++) {
          const float e = __expf(sc[nt][r] - nm);
          sc[nt][r] = e;
          ps += e;
        }
#pragma unroll
        for (int off = 1; off < 16; off <<= 1)
          ps += __shfl_xor(ps, off, 32);
        rowsum[r] = rowsum[r] * corr + ps;
        rowmax[r] = nm;
#pragma unroll
        for (int nt = 0; nt < 4; nt++) yacc[nt][r] *= corr;
      }
      // --- relayout P: D-frag -> row-major LDS -> A-frag (same-wave, in order)
#pragma unroll
      for (int nt = 0; nt < 2; nt++)
#pragma unroll
        for (int r = 0; r < 8; r++)
          Plds[wave][(r + half * 8) * 32 + nt * 16 + ln] = (_Float16)sc[nt][r];
      V16 pa;
      {
        const _Float16* p = &Plds[wave][ln * 32 + half * 8];
        pa.h[0] = *(const v8h*)p;         // kk = half*8      .. +7
        pa.h[1] = *(const v8h*)(p + 16);  // kk = 16 + half*8 .. +7
      }
      // --- y += P @ V  (B[kk][n] = Vt[d][kk], contiguous kk) ---
#pragma unroll
      for (int nt = 0; nt < 4; nt++) {
        V16 bf;
        const _Float16* p = Vb + (nt * 16 + ln) * 32 + half * 8;
        bf.h[0] = *(const v8h*)p;
        bf.h[1] = *(const v8h*)(p + 16);
        yacc[nt] = wmma_f16(pa.v, bf.v, yacc[nt]);
      }
    }
  }

  // --- epilogue: y = yacc / rowsum ---
  const size_t obase = ((size_t)b * SEQ + qlo) * DM;
#pragma unroll
  for (int r = 0; r < 8; r++) {
    const float inv = 1.0f / rowsum[r];
    const int mrow = r + half * 8;
#pragma unroll
    for (int nt = 0; nt < 4; nt++)
      out[obase + (size_t)mrow * DM + nt * 16 + ln] = yacc[nt][r] * inv;
  }
}

// ===========================================================================
// Launch: proj (1024 WGs) then attention (1024 WGs), both on `stream`.
// Workspace: f16 Q, K, V -> 3 * 16*4096*64 * 2 B = 24 MB in d_ws.
// ===========================================================================
extern "C" void kernel_launch(void* const* d_in, const int* in_sizes, int n_in,
                              void* d_out, int out_size, void* d_ws, size_t ws_size,
                              hipStream_t stream) {
  const float* x  = (const float*)d_in[0];
  const float* Wk = (const float*)d_in[1];
  const float* Wq = (const float*)d_in[2];
  const float* Wv = (const float*)d_in[3];

  const size_t elems = (size_t)BATCH * SEQ * DM;
  _Float16* Qh = (_Float16*)d_ws;
  _Float16* Kh = Qh + elems;
  _Float16* Vh = Kh + elems;

  dim3 block(128);
  dim3 gridProj(BATCH * SEQ / 64);   // 1024
  dim3 gridAttn(BATCH * (SEQ / 64)); // 1024

  qkv_proj_kernel<<<gridProj, block, 0, stream>>>(x, Wk, Wq, Wv, Qh, Kh, Vh);
  attn_kernel<<<gridAttn, block, 0, stream>>>(Qh, Kh, Vh, (float*)d_out);
}
